// Net_75282186764473
// MI455X (gfx1250) — matrix operand order repaired and checked by reading, not compiled, not run on hardware
//
#include <hip/hip_runtime.h>

// ---------------------------------------------------------------------------
// PAT ODE network forward for MI455X (gfx1250), fully fused:
//   stage s: 5 explicit-Euler steps of  q' = p, p' = e + sin(1.1 q) @ A_s
//   with A_s = c2q(W_s) + Qn_s - I  (the forward value of pat() is just the
//   'experiment' trajectory; stop_gradient only affects backward).
// Effective matrices precomputed to f16 in d_ws (transposed, padded 224x232),
// staged through LDS; state q,p kept in WMMA C-fragment registers; all GEMM
// work on v_wmma_f32_16x16x32_f16.
// ---------------------------------------------------------------------------

typedef _Float16 v16h __attribute__((ext_vector_type(16)));
typedef _Float16 v8h  __attribute__((ext_vector_type(8)));
typedef float    v8f  __attribute__((ext_vector_type(8)));

#define D1      196
#define D2      206
#define DPAD    224          // padded dimension: 14*16 = 7*32
#define WSTR    232          // f16 row stride (bank-conflict-free B128 loads)
#define NTILE   14           // N tiles of 16
#define KTILE   7            // K tiles of 32
#define STEPS   5
#define DTSTEP  0.1f
#define ROWS_PER_BLOCK 128   // 8 waves * 16 batch rows

// Build one 16-element f16 fragment (8 VGPRs) from two aligned 16B LDS loads.
__device__ __forceinline__ v16h ld_frag16(const _Float16* p0, const _Float16* p1) {
  v8h lo = *(const v8h*)p0;
  v8h hi = *(const v8h*)p1;
  v16h r;
#pragma unroll
  for (int i = 0; i < 8; ++i) { r[i] = lo[i]; r[i + 8] = hi[i]; }
  return r;
}

// ---------------------------------------------------------------------------
// Prep: outAT[j][k] = (f16)( A[k][j] ), A = c2q(W) + Qn - I, zero-padded.
// One block per output row j (0..DPAD-1).
// ---------------------------------------------------------------------------
__global__ void prep_kernel(const float* __restrict__ W,
                            const float* __restrict__ Qn,
                            _Float16* __restrict__ outAT,
                            int D) {
  __shared__ float red[256];
  const int j = blockIdx.x;
  const int t = threadIdx.x;

  float partial = 0.0f;
  if (j < D) {
    for (int i = t; i < D; i += 256)
      partial += 0.5f * (W[(size_t)i * D + j] + W[(size_t)j * D + i]);
  }
  red[t] = partial;
  __syncthreads();
#pragma unroll
  for (int s = 128; s > 0; s >>= 1) {
    if (t < s) red[t] += red[t + s];
    __syncthreads();
  }
  const float diag = -red[0];   // c2q diagonal: -sum of symmetrized column

  if (t < WSTR) {
    const int k = t;
    float val = 0.0f;
    if (j < D && k < D) {
      float qs = (k == j) ? diag
                          : 0.5f * (W[(size_t)k * D + j] + W[(size_t)j * D + k]);
      val = qs + Qn[(size_t)k * D + j];
      if (k == j) val -= 1.0f;  // folds the "-sq" term into the matrix
    }
    outAT[(size_t)j * WSTR + k] = (_Float16)val;
  }
}

// ---------------------------------------------------------------------------
// Main fused kernel: 256 threads (8 wave32), 128 batch rows per block.
// ---------------------------------------------------------------------------
__global__ __launch_bounds__(256)
void pat_main(const float* __restrict__ x,
              const float* __restrict__ bias1,
              const float* __restrict__ bias2,
              const float* __restrict__ ofac,
              const _Float16* __restrict__ wAT,   // 2 stages of [DPAD][WSTR] f16
              float* __restrict__ out) {
  extern __shared__ char smem[];
  _Float16* ldsW = (_Float16*)smem;                          // [DPAD][WSTR]
  const int tid  = threadIdx.x;
  const int wave = tid >> 5;
  const int lane = tid & 31;
  const int hi   = lane >> 4;      // half-wave select
  const int nlo  = lane & 15;
  const int mhi  = hi ? 8 : 0;
  _Float16* sqBuf = ldsW + DPAD * WSTR + wave * (16 * WSTR); // per-wave [16][WSTR]

  const long rowBase = (long)blockIdx.x * ROWS_PER_BLOCK + wave * 16;

  // State in WMMA C-fragment layout: element (VGPR v, lane) = (M=v+mhi, N=nlo).
  v8f q[NTILE], p[NTILE];
#pragma unroll
  for (int nt = 0; nt < NTILE; ++nt) {
    const int col = nt * 16 + nlo;
#pragma unroll
    for (int v = 0; v < 8; ++v) {
      const long row = rowBase + v + mhi;
      q[nt][v] = (col < D1) ? x[row * D1 + col] : 0.0f;
      p[nt][v] = 0.0f;
    }
  }

  float ev[NTILE];

#pragma unroll 1
  for (int stage = 0; stage < 2; ++stage) {
    __syncthreads();  // all waves done reading previous-stage weights
    {   // cooperative weight load: global(ws) -> LDS, 16B chunks
      const uint4* src = (const uint4*)(wAT + (size_t)stage * DPAD * WSTR);
      uint4* dst = (uint4*)ldsW;
      const int n16 = DPAD * WSTR * (int)sizeof(_Float16) / 16;
#pragma unroll 1
      for (int i = tid; i < n16; i += 256) dst[i] = src[i];
    }
    __syncthreads();

    const float* bias = stage ? bias2 : bias1;
    const int D = stage ? D2 : D1;
#pragma unroll
    for (int nt = 0; nt < NTILE; ++nt) {
      const int col = nt * 16 + nlo;
      ev[nt] = (col < D) ? bias[col] : 0.0f;
    }
    if (stage) {  // q carries over (pad cols stay exactly 0), momentum resets
#pragma unroll
      for (int nt = 0; nt < NTILE; ++nt)
#pragma unroll
        for (int v = 0; v < 8; ++v) p[nt][v] = 0.0f;
    }

#pragma unroll 1
    for (int step = 0; step < STEPS; ++step) {
      // sin(1.1*q) -> f16 per-wave LDS tile (C-layout scatter; same-wave LDS
      // ops are in-order, so no barrier is required before the reloads).
#pragma unroll
      for (int nt = 0; nt < NTILE; ++nt) {
#pragma unroll
        for (int v = 0; v < 8; ++v) {
          sqBuf[(v + mhi) * WSTR + nt * 16 + nlo] =
              (_Float16)__sinf(1.1f * q[nt][v]);
        }
      }

      // A fragments (16x32 f16): lanes 0-15 row M=nlo K{0..7,16..23},
      // lanes 16-31 row M=nlo K{8..15,24..31}.
      v16h afrag[KTILE];
      const _Float16* ar = sqBuf + nlo * WSTR + (hi ? 8 : 0);
#pragma unroll
      for (int kt = 0; kt < KTILE; ++kt)
        afrag[kt] = ld_frag16(ar + kt * 32, ar + kt * 32 + 16);

#pragma unroll
      for (int nt = 0; nt < NTILE; ++nt) {
        v8f acc;
#pragma unroll
        for (int v = 0; v < 8; ++v) acc[v] = ev[nt];  // bias as C matrix
        // B fragments (32x16 f16): lane = column N, K{0..7,16..23} / K{8..15,24..31} from A^T rows
        const _Float16* br = ldsW + (nt * 16 + nlo) * WSTR + (hi ? 8 : 0);
#pragma unroll
        for (int kt = 0; kt < KTILE; ++kt) {
          v16h bfrag = ld_frag16(br + kt * 32, br + kt * 32 + 16);
          acc = __builtin_amdgcn_wmma_f32_16x16x32_f16(
              false, afrag[kt], false, bfrag, (short)0, acc, false, false);
        }
        // Euler: q uses old p (update q first), p uses dp(old q)
#pragma unroll
        for (int v = 0; v < 8; ++v) {
          q[nt][v] += DTSTEP * p[nt][v];
          p[nt][v] += DTSTEP * acc[v];
        }
      }
    }
  }

  // Output: columns 196..205 live in N-tile 12 (cols 192..207)
  const float f = ofac[0];
  const int col = 12 * 16 + nlo;
  if (col >= D1 && col < D2) {
#pragma unroll
    for (int v = 0; v < 8; ++v)
      out[(rowBase + v + mhi) * 10 + (col - D1)] = f * q[12][v];
  }
}

// ---------------------------------------------------------------------------
extern "C" void kernel_launch(void* const* d_in, const int* in_sizes, int n_in,
                              void* d_out, int out_size, void* d_ws, size_t ws_size,
                              hipStream_t stream) {
  (void)n_in; (void)out_size; (void)ws_size;
  const float* x    = (const float*)d_in[0];
  const float* w1   = (const float*)d_in[1];
  const float* b1   = (const float*)d_in[2];
  const float* w2   = (const float*)d_in[3];
  const float* b2   = (const float*)d_in[4];
  const float* ofac = (const float*)d_in[5];
  const float* qn1  = (const float*)d_in[6];
  const float* qn2  = (const float*)d_in[7];
  _Float16* ws16 = (_Float16*)d_ws;   // 2 * DPAD * WSTR f16 = ~208 KB
  float* out = (float*)d_out;

  const int B = in_sizes[0] / D1;     // 65536

  prep_kernel<<<dim3(DPAD), dim3(256), 0, stream>>>(w1, qn1, ws16, D1);
  prep_kernel<<<dim3(DPAD), dim3(256), 0, stream>>>(w2, qn2,
                                                    ws16 + DPAD * WSTR, D2);

  const size_t shmem = (size_t)(DPAD * WSTR + 8 * 16 * WSTR) * sizeof(_Float16);
  hipFuncSetAttribute(reinterpret_cast<const void*>(pat_main),
                      hipFuncAttributeMaxDynamicSharedMemorySize, (int)shmem);
  pat_main<<<dim3(B / ROWS_PER_BLOCK), dim3(256), shmem, stream>>>(
      x, b1, b2, ofac, ws16, out);
}
